// AttentionHead_38732015075596
// MI455X (gfx1250) — matrix operand order repaired and checked by reading, not compiled, not run on hardware
//
#include <hip/hip_runtime.h>
#include <hip/hip_bf16.h>

typedef __attribute__((ext_vector_type(16))) _Float16 v16h;
typedef __attribute__((ext_vector_type(8)))  _Float16 v8h;
typedef __attribute__((ext_vector_type(8)))  float    v8f;

#define D_IN 256
#define SEQ  2048
#define DH   64
#define NB   16
#define LOG2E 1.4426950408889634f

static __device__ __forceinline__ v16h vcat(v8h a, v8h b) {
  return __builtin_shufflevector(a, b, 0,1,2,3,4,5,6,7,8,9,10,11,12,13,14,15);
}

// A-fragment (16x32 f16, M = lane%16): halfs [g*8 .. +7] and [16+g*8 .. +7]
static __device__ __forceinline__ v16h load_afrag(const _Float16* row, int g) {
  v8h lo = *(const v8h*)(row + g * 8);
  v8h hi = *(const v8h*)(row + 16 + g * 8);
  return vcat(lo, hi);
}

// B-fragment (32x16 f16, N = lane%16): halfs [g*8 .. +7] and [16+g*8 .. +7]
static __device__ __forceinline__ v16h load_bfrag(const _Float16* p) {
  v8h lo = *(const v8h*)(p);
  v8h hi = *(const v8h*)(p + 16);
  return vcat(lo, hi);
}

static __device__ __forceinline__ v8f wmma_f16(v16h a, v16h b, v8f c) {
  return __builtin_amdgcn_wmma_f32_16x16x32_f16(false, a, false, b, (short)0, c,
                                                false, false);
}

// ---------------------------------------------------------------------------
// Kernel 0: one-shot weight convert+transpose. w[m][k][n] f32 -> wT[m][n][k] f16
// ---------------------------------------------------------------------------
__global__ __launch_bounds__(256) void wconv_kernel(
    const float* __restrict__ wq, const float* __restrict__ wk,
    const float* __restrict__ wv, _Float16* __restrict__ wT) {
  int idx = blockIdx.x * 256 + threadIdx.x;      // 3*64*256 = 49152 elements
  if (idx >= 3 * DH * D_IN) return;
  int m   = idx / (DH * D_IN);
  int rem = idx - m * (DH * D_IN);
  int n   = rem / D_IN;
  int k   = rem - n * D_IN;
  const float* w = (m == 0) ? wq : ((m == 1) ? wk : wv);
  wT[idx] = (_Float16)w[(size_t)k * DH + n];
}

// ---------------------------------------------------------------------------
// Kernel 1: QKV projection, barrier-free. All 8 A-fragments of this lane's
// x row are loaded+converted once up front; Q, K, V are then processed
// sequentially so only 4 accumulators + 4 distinct B-fragment registers are
// live, letting the 8 weight loads per K-step issue as one clause ahead of
// the 4 WMMAs. q gets bias + 1/sqrt(64) folded in.
// Outputs: qh,kh [B,N,64] f16 row-major; vT [B,64,N] f16 (transposed).
// ---------------------------------------------------------------------------
__global__ __launch_bounds__(256) void qkv_proj_kernel(
    const float* __restrict__ x, const _Float16* __restrict__ wT,
    const float* __restrict__ bq, const float* __restrict__ bk,
    const float* __restrict__ bv,
    _Float16* __restrict__ qh, _Float16* __restrict__ kh,
    _Float16* __restrict__ vT) {
  const int b       = blockIdx.x >> 4;
  const int rowBase = (blockIdx.x & 15) * 128;
  const int wave = threadIdx.x >> 5;
  const int lane = threadIdx.x & 31;
  const int g    = lane >> 4;
  const int lm   = lane & 15;

  const float* xrow = x + ((size_t)b * SEQ + rowBase + wave * 16 + lm) * D_IN;

  // --- load + convert all A-fragments (this lane's whole x row, once) ---
  v16h af[8];
#pragma unroll
  for (int kk = 0; kk < 8; ++kk) {
    const float* p0 = xrow + kk * 32 + g * 8;        // e=0..7  (K=g*8+e)
    const float4 a0 = *(const float4*)(p0);
    const float4 a1 = *(const float4*)(p0 + 4);
    const float4 a2 = *(const float4*)(p0 + 16);     // e=8..15 (K=16+g*8+..)
    const float4 a3 = *(const float4*)(p0 + 20);
    v16h a;
    a[0]=(_Float16)a0.x; a[1]=(_Float16)a0.y; a[2]=(_Float16)a0.z; a[3]=(_Float16)a0.w;
    a[4]=(_Float16)a1.x; a[5]=(_Float16)a1.y; a[6]=(_Float16)a1.z; a[7]=(_Float16)a1.w;
    a[8]=(_Float16)a2.x; a[9]=(_Float16)a2.y; a[10]=(_Float16)a2.z; a[11]=(_Float16)a2.w;
    a[12]=(_Float16)a3.x; a[13]=(_Float16)a3.y; a[14]=(_Float16)a3.z; a[15]=(_Float16)a3.w;
    af[kk] = a;
  }

  const int rowTop = rowBase + wave * 16 + g * 8;    // C layout: M = r + 8*g

#pragma unroll
  for (int m = 0; m < 3; ++m) {
    v8f acc[4];
#pragma unroll
    for (int nt = 0; nt < 4; ++nt) acc[nt] = (v8f)(0.f);
    const _Float16* wbase = wT + (size_t)m * DH * D_IN;

#pragma unroll
    for (int kk = 0; kk < 8; ++kk) {
      v16h bf[4];
#pragma unroll
      for (int nt = 0; nt < 4; ++nt)
        bf[nt] = load_bfrag(wbase + (size_t)(nt * 16 + lm) * D_IN + kk * 32 + g * 8);
#pragma unroll
      for (int nt = 0; nt < 4; ++nt)
        acc[nt] = wmma_f16(af[kk], bf[nt], acc[nt]);
    }

    // per-matrix epilogue retires accumulators immediately
#pragma unroll
    for (int nt = 0; nt < 4; ++nt) {
      const int n = nt * 16 + lm;
      if (m == 0) {
        const float bb = bq[n];
#pragma unroll
        for (int r = 0; r < 8; ++r)
          qh[((size_t)b * SEQ + rowTop + r) * DH + n] =
              (_Float16)((acc[nt][r] + bb) * 0.125f);
      } else if (m == 1) {
        const float bb = bk[n];
#pragma unroll
        for (int r = 0; r < 8; ++r)
          kh[((size_t)b * SEQ + rowTop + r) * DH + n] =
              (_Float16)(acc[nt][r] + bb);
      } else {
        const float bb = bv[n];
        v8h pv;
#pragma unroll
        for (int r = 0; r < 8; ++r) pv[r] = (_Float16)(acc[nt][r] + bb);
        *(v8h*)(vT + ((size_t)b * DH + n) * SEQ + rowTop) = pv;
      }
    }
  }
}

// ---------------------------------------------------------------------------
// Kernel 2: flash attention, 64-key tiles, double-buffered LDS staging.
// 128 query rows per block (8 waves x 16 rows), online softmax.
// ---------------------------------------------------------------------------
__global__ __launch_bounds__(256) void flash_attn_kernel(
    const _Float16* __restrict__ qh, const _Float16* __restrict__ kh,
    const _Float16* __restrict__ vT, float* __restrict__ out) {
  __shared__ _Float16 ks[2][64 * 72];       // K tile  [key][64], pad 72
  __shared__ _Float16 vs[2][64 * 72];       // V^T tile [dim][64], pad 72
  __shared__ _Float16 ps[8][16 * 72];       // per-wave P tile [16][64], pad 72

  const int b       = blockIdx.x >> 4;
  const int rowBase = (blockIdx.x & 15) * 128;
  const int t    = threadIdx.x;
  const int wave = t >> 5;
  const int lane = t & 31;
  const int g    = lane >> 4;
  const int lm   = lane & 15;

  // staging map: each thread moves rows j and j+32 (chunk ch) of both tiles
  const int j  = t >> 3;
  const int ch = (t & 7) * 8;
  const _Float16* kbase = kh + ((size_t)b * SEQ) * DH;
  const _Float16* vbase = vT + ((size_t)b * DH) * SEQ;

  // Q A-fragments (row M = lm), 2 K-steps over d=64
  const _Float16* qrow = qh + ((size_t)b * SEQ + rowBase + wave * 16 + lm) * DH;
  const v16h qa0 = load_afrag(qrow, g);
  const v16h qa1 = load_afrag(qrow + 32, g);

  v8f o[4];
#pragma unroll
  for (int nt = 0; nt < 4; ++nt) o[nt] = (v8f)(0.f);
  float mrow[8], lrow[8];
#pragma unroll
  for (int r = 0; r < 8; ++r) { mrow[r] = -3.0e38f; lrow[r] = 0.f; }

  // prologue: stage tile 0 into buffer 0
  {
    *(v8h*)(&ks[0][j * 72 + ch])        = *(const v8h*)(kbase + (size_t)j * DH + ch);
    *(v8h*)(&ks[0][(j + 32) * 72 + ch]) = *(const v8h*)(kbase + (size_t)(j + 32) * DH + ch);
    *(v8h*)(&vs[0][j * 72 + ch])        = *(const v8h*)(vbase + (size_t)j * SEQ + ch);
    *(v8h*)(&vs[0][(j + 32) * 72 + ch]) = *(const v8h*)(vbase + (size_t)(j + 32) * SEQ + ch);
  }
  __syncthreads();

  for (int jt = 0; jt < SEQ / 64; ++jt) {
    const int cur = jt & 1;
    const bool have_next = (jt + 1) < (SEQ / 64);

    // issue next tile's global loads early; ds stores land after compute
    v8h nk0, nk1, nv0, nv1;
    if (have_next) {
      const size_t kb = (size_t)(jt + 1) * 64;
      nk0 = *(const v8h*)(kbase + (kb + j) * DH + ch);
      nk1 = *(const v8h*)(kbase + (kb + j + 32) * DH + ch);
      nv0 = *(const v8h*)(vbase + (size_t)j * SEQ + kb + ch);
      nv1 = *(const v8h*)(vbase + (size_t)(j + 32) * SEQ + kb + ch);
    }

    // --- S = Q * K^T : 4 column tiles x 2 K-steps = 8 WMMAs ---
    v8f s[4];
#pragma unroll
    for (int jc = 0; jc < 4; ++jc) {
      s[jc] = (v8f)(0.f);
      const _Float16* krow = &ks[cur][(jc * 16 + lm) * 72 + g * 8];
      s[jc] = wmma_f16(qa0, load_bfrag(krow), s[jc]);
      s[jc] = wmma_f16(qa1, load_bfrag(krow + 32), s[jc]);
    }

    // --- online softmax (rows M = g*8 + r; reduce across 16 lanes) ---
    float corr[8];
#pragma unroll
    for (int r = 0; r < 8; ++r) {
      float mx = fmaxf(fmaxf(s[0][r], s[1][r]), fmaxf(s[2][r], s[3][r]));
      mx = fmaxf(mx, __shfl_xor(mx, 1));
      mx = fmaxf(mx, __shfl_xor(mx, 2));
      mx = fmaxf(mx, __shfl_xor(mx, 4));
      mx = fmaxf(mx, __shfl_xor(mx, 8));
      const float mn = fmaxf(mrow[r], mx);
      const float c  = exp2f((mrow[r] - mn) * LOG2E);
      mrow[r] = mn;
      corr[r] = c;
      float p[4], rs = 0.f;
#pragma unroll
      for (int jc = 0; jc < 4; ++jc) {
        p[jc] = exp2f((s[jc][r] - mn) * LOG2E);
        rs += p[jc];
      }
      rs += __shfl_xor(rs, 1);
      rs += __shfl_xor(rs, 2);
      rs += __shfl_xor(rs, 4);
      rs += __shfl_xor(rs, 8);
      lrow[r] = lrow[r] * c + rs;
      _Float16* prow_w = &ps[wave][(g * 8 + r) * 72 + lm];
#pragma unroll
      for (int jc = 0; jc < 4; ++jc) prow_w[jc * 16] = (_Float16)p[jc];
    }
#pragma unroll
    for (int nt = 0; nt < 4; ++nt)
#pragma unroll
      for (int r = 0; r < 8; ++r) o[nt][r] *= corr[r];

    // same-wave LDS RAW on P tile: wait on CDNA5 split DS counter
    asm volatile("s_wait_dscnt 0x0" ::: "memory");

    // --- O += P * V : 2 K-steps x 4 N-tiles = 8 WMMAs ---
    const _Float16* prow = &ps[wave][lm * 72];
    const v16h pa0 = load_afrag(prow, g);
    const v16h pa1 = load_afrag(prow + 32, g);
#pragma unroll
    for (int nt = 0; nt < 4; ++nt) {
      const _Float16* vrow = &vs[cur][(nt * 16 + lm) * 72 + g * 8];
      o[nt] = wmma_f16(pa0, load_bfrag(vrow), o[nt]);
      o[nt] = wmma_f16(pa1, load_bfrag(vrow + 32), o[nt]);
    }

    // land the staged next tile into the other buffer, then flip
    if (have_next) {
      const int nxt = cur ^ 1;
      *(v8h*)(&ks[nxt][j * 72 + ch])        = nk0;
      *(v8h*)(&ks[nxt][(j + 32) * 72 + ch]) = nk1;
      *(v8h*)(&vs[nxt][j * 72 + ch])        = nv0;
      *(v8h*)(&vs[nxt][(j + 32) * 72 + ch]) = nv1;
    }
    __syncthreads();
  }

  // --- final normalize + store f32 output ---
  const int rowTop = rowBase + wave * 16 + g * 8;
#pragma unroll
  for (int nt = 0; nt < 4; ++nt) {
    const int n = nt * 16 + lm;
#pragma unroll
    for (int r = 0; r < 8; ++r) {
      out[((size_t)b * SEQ + rowTop + r) * DH + n] = o[nt][r] / lrow[r];
    }
  }
}

extern "C" void kernel_launch(void* const* d_in, const int* in_sizes, int n_in,
                              void* d_out, int out_size, void* d_ws, size_t ws_size,
                              hipStream_t stream) {
  const float* x  = (const float*)d_in[0];
  const float* wq = (const float*)d_in[1];
  const float* bq = (const float*)d_in[2];
  const float* wk = (const float*)d_in[3];
  const float* bk = (const float*)d_in[4];
  const float* wv = (const float*)d_in[5];
  const float* bv = (const float*)d_in[6];
  float* out = (float*)d_out;

  _Float16* qh = (_Float16*)d_ws;
  _Float16* kh = qh + (size_t)NB * SEQ * DH;
  _Float16* vT = kh + (size_t)NB * SEQ * DH;
  _Float16* wT = vT + (size_t)NB * SEQ * DH;      // 3*64*256 f16

  dim3 block(256);
  wconv_kernel<<<dim3((3 * DH * D_IN + 255) / 256), block, 0, stream>>>(wq, wk, wv, wT);
  qkv_proj_kernel<<<dim3(NB * (SEQ / 128)), block, 0, stream>>>(x, wT, bq, bk, bv,
                                                                qh, kh, vT);
  flash_attn_kernel<<<dim3(NB * (SEQ / 128)), block, 0, stream>>>(qh, kh, vT, out);
}
